// MECM_62285615726967
// MI455X (gfx1250) — matrix operand order repaired and checked by reading, not compiled, not run on hardware
//
#include <hip/hip_runtime.h>

// ---------------- types ----------------
typedef __bf16 bf16_t;
typedef __attribute__((ext_vector_type(16))) __bf16 v16bf;
typedef __attribute__((ext_vector_type(8)))  __bf16 v8bf;
typedef __attribute__((ext_vector_type(8)))  float  v8f;

// ---------------- problem constants ----------------
#define VOCAB   32000
#define EMB     43
#define G4      (4*EMB)      // 172 gate units
#define LAYERS  64
#define OUT_DIM 15
#define NTOK    524288

// padded GEMM shape per layer: (16 x 64) x (64 x 176)
#define KPAD    64
#define NPAD    176
#define NTILES  (NPAD/16)    // 11
#define WSLAB_V4 ((KPAD*NPAD)/8)   // 1408 uint4 per layer slab
#define WREGS    6                 // ceil(1408/256)

// workgroup shape
#define WAVES   8
#define TPW     16                 // tokens per wave (WMMA M)
#define TPB     (WAVES*TPW)        // 128 tokens per workgroup
#define NBLOCKS (NTOK/TPB)         // 4096

// workspace layout (bytes)
#define WT_BYTES   ((size_t)LAYERS*KPAD*NPAD*2)   // 1,441,792

// ---------------- fast activations ----------------
__device__ __forceinline__ float fast_sigmoid(float x) {
    return __builtin_amdgcn_rcpf(1.0f + __expf(-x));
}
__device__ __forceinline__ float fast_tanh(float x) {
    return 2.0f * __builtin_amdgcn_rcpf(1.0f + __expf(-2.0f * x)) - 1.0f;
}

// ---------------- prep: transpose+pad w_ih to bf16, fuse biases ----------------
__global__ void prep_weights(const float* __restrict__ wih,
                             const float* __restrict__ bih,
                             const float* __restrict__ bhh,
                             bf16_t* __restrict__ wt,
                             float* __restrict__ bsum) {
    const int l = blockIdx.x;
    const float* W = wih + (size_t)l * G4 * EMB;       // (172, 43) row-major
    bf16_t* w = wt + (size_t)l * KPAD * NPAD;          // (176, 64) n-major
    for (int idx = threadIdx.x; idx < KPAD * NPAD; idx += blockDim.x) {
        int k = idx / NPAD;
        int n = idx - k * NPAD;
        float v = (k < EMB && n < G4) ? W[n * EMB + k] : 0.0f;
        w[n * KPAD + k] = (bf16_t)v;
    }
    float* bs = bsum + (size_t)l * NPAD;
    for (int n = threadIdx.x; n < NPAD; n += blockDim.x)
        bs[n] = (n < G4) ? (bih[l * G4 + n] + bhh[l * G4 + n]) : 0.0f;
}

// ---------------- main: 64-layer LSTM stack + projection + log_softmax ----------------
__global__ __launch_bounds__(256) void lstm_stack(
        const int*    __restrict__ tokens,
        const float*  __restrict__ emb,
        const bf16_t* __restrict__ wt,
        const float*  __restrict__ bsum,
        const float*  __restrict__ w_out,
        const float*  __restrict__ b_out,
        float*        __restrict__ out) {

    __shared__ __align__(16) bf16_t sWt[2][KPAD * NPAD];     // 2 x 22528 B ping-pong
    __shared__ float  sBias[2][NPAD];                        // fused bias ping-pong
    __shared__ float  sWout[OUT_DIM * EMB];                  // output proj
    __shared__ float  sBout[OUT_DIM];
    __shared__ int    sTok[TPB];
    __shared__ __align__(16) bf16_t sH[WAVES][TPW * KPAD];   // per-wave h (A staging)
    __shared__ float  sG[WAVES][TPW * NPAD];                 // per-wave gates / logits

    const int tid  = threadIdx.x;
    const int w    = tid >> 5;
    const int lane = tid & 31;
    const int half = lane >> 4;     // 0: lanes 0-15, 1: lanes 16-31
    const int l16  = lane & 15;
    const long base = (long)blockIdx.x * TPB;

    // stage constants + layer-0 weight slab (b128 wide)
    for (int i = tid; i < OUT_DIM * EMB; i += 256) sWout[i] = w_out[i];
    if (tid < OUT_DIM) sBout[tid] = b_out[tid];
    for (int i = tid; i < TPB; i += 256) sTok[i] = tokens[base + i];
    {
        const uint4* src = (const uint4*)wt;
        uint4* dst = (uint4*)sWt[0];
        for (int i = tid; i < WSLAB_V4; i += 256) dst[i] = src[i];
        if (tid < NPAD) sBias[0][tid] = bsum[tid];
    }
    __syncthreads();

    bf16_t* hb = sH[w];
    float*  gb = sG[w];

    // h0 = embedding gather, zero-padded to K=64 (per wave, no sync needed)
    for (int i = lane; i < TPW * KPAD; i += 32) hb[i] = (bf16_t)0.0f;
    for (int i = lane; i < TPW * EMB; i += 32) {
        int m = i / EMB;
        int u = i - m * EMB;
        int t = sTok[w * TPW + m];
        hb[m * KPAD + u] = (bf16_t)emb[(size_t)t * EMB + u];
    }

    for (int l = 0; l < LAYERS; ++l) {
        const int p = l & 1;

        // issue next layer's slab into registers early (latency hidden under WMMA)
        uint4 wreg[WREGS];
        float breg = 0.0f;
        if (l + 1 < LAYERS) {
            const uint4* src = (const uint4*)(wt + (size_t)(l + 1) * KPAD * NPAD);
#pragma unroll
            for (int j = 0; j < WREGS; ++j) {
                int i = tid + j * 256;
                if (i < WSLAB_V4) wreg[j] = src[i];
            }
            if (tid < NPAD) breg = bsum[(size_t)(l + 1) * NPAD + tid];
            if (l + 2 < LAYERS) {   // global_prefetch_b8 one more layer ahead
                const char* nxt = (const char*)(wt + (size_t)(l + 2) * KPAD * NPAD);
                __builtin_prefetch(nxt + tid * 88, 0, 1);
            }
        }

        const bf16_t* wl = sWt[p];
        const float*  bl = sBias[p];

        // A fragments (16x32 bf16, ISA 7.12.2 layout): lane<16 holds K 0-7,16-23 of row l16;
        // lane>=16 holds K 8-15,24-31. ua1 is the K=32..63 half.
        union { v16bf v; v8bf h[2]; } ua0, ua1;
        {
            const int rb = l16 * KPAD + 8 * half;
            ua0.h[0] = *(const v8bf*)&hb[rb];
            ua0.h[1] = *(const v8bf*)&hb[rb + 16];
            ua1.h[0] = *(const v8bf*)&hb[rb + 32];
            ua1.h[1] = *(const v8bf*)&hb[rb + 48];
        }

        for (int nt = 0; nt < NTILES; ++nt) {
            // B fragments (32x16 bf16): lane holds column N=l16, K in same order as A (b1: +32)
            union { v16bf v; v8bf h[2]; } ub0, ub1;
            const int off0 = (nt * 16 + l16) * KPAD + 8 * half;
            const int off1 = off0 + 32;
            ub0.h[0] = *(const v8bf*)&wl[off0];
            ub0.h[1] = *(const v8bf*)&wl[off0 + 16];
            ub1.h[0] = *(const v8bf*)&wl[off1];
            ub1.h[1] = *(const v8bf*)&wl[off1 + 16];

            v8f acc = {0.f, 0.f, 0.f, 0.f, 0.f, 0.f, 0.f, 0.f};
            acc = __builtin_amdgcn_wmma_f32_16x16x32_bf16(
                    false, ua0.v, false, ub0.v, (short)0, acc, false, false);
            acc = __builtin_amdgcn_wmma_f32_16x16x32_bf16(
                    false, ua1.v, false, ub1.v, (short)0, acc, false, false);

            // scatter accumulator to gates with fused bias:
            // VGPR r -> row M = r + 8*half, col N = nt*16 + l16
            const int n = nt * 16 + l16;
            const float bn = bl[n];
#pragma unroll
            for (int r = 0; r < 8; ++r)
                gb[(r + 8 * half) * NPAD + n] = acc[r] + bn;
        }

        // activations: c = sig(i)*tanh(g); h = sig(o)*tanh(c)   (f-gate dead, c_prev=0)
        for (int i = lane; i < TPW * EMB; i += 32) {
            int m = i / EMB;
            int u = i - m * EMB;
            const float* g = gb + m * NPAD;
            float gi = g[u];
            float gg = g[2 * EMB + u];
            float go = g[3 * EMB + u];
            float c  = fast_sigmoid(gi) * fast_tanh(gg);
            float hn = fast_sigmoid(go) * fast_tanh(c);
            hb[m * KPAD + u] = (bf16_t)hn;
        }

        // drain next layer's slab into the alternate LDS buffer
        if (l + 1 < LAYERS) {
            uint4* dst = (uint4*)sWt[1 - p];
#pragma unroll
            for (int j = 0; j < WREGS; ++j) {
                int i = tid + j * 256;
                if (i < WSLAB_V4) dst[i] = wreg[j];
            }
            if (tid < NPAD) sBias[1 - p][tid] = breg;
        }
        __syncthreads();   // buf (1-p) filled; everyone done reading buf p
    }

    // output projection: logits[m][o] = b_out[o] + h[m] . w_out[o]
    for (int i = lane; i < TPW * OUT_DIM; i += 32) {
        int m = i / OUT_DIM;
        int o = i - m * OUT_DIM;
        float s = sBout[o];
        const bf16_t* hr = hb + m * KPAD;
        const float*  wr = sWout + o * EMB;
        for (int k = 0; k < EMB; ++k) s += (float)hr[k] * wr[k];
        gb[m * NPAD + o] = s;
    }

    // log_softmax over 15 classes, one token per lane
    if (lane < TPW) {
        const int m = lane;
        const float* lg = gb + m * NPAD;
        float mx = lg[0];
#pragma unroll
        for (int o = 1; o < OUT_DIM; ++o) mx = fmaxf(mx, lg[o]);
        float se = 0.0f;
#pragma unroll
        for (int o = 0; o < OUT_DIM; ++o) se += __expf(lg[o] - mx);
        const float lz = mx + __logf(se);
        float* po = out + (base + (long)w * TPW + m) * OUT_DIM;
#pragma unroll
        for (int o = 0; o < OUT_DIM; ++o) po[o] = lg[o] - lz;
    }
}

// ---------------- launcher ----------------
extern "C" void kernel_launch(void* const* d_in, const int* in_sizes, int n_in,
                              void* d_out, int out_size, void* d_ws, size_t ws_size,
                              hipStream_t stream) {
    const int*   tokens = (const int*)  d_in[0];
    const float* emb    = (const float*)d_in[1];
    const float* w_ih   = (const float*)d_in[2];
    // d_in[3] = w_hh: mathematically dead (h_prev = 0)
    const float* b_ih   = (const float*)d_in[4];
    const float* b_hh   = (const float*)d_in[5];
    const float* w_out  = (const float*)d_in[6];
    const float* b_out  = (const float*)d_in[7];
    float* out = (float*)d_out;

    bf16_t* wt   = (bf16_t*)d_ws;
    float*  bsum = (float*)((char*)d_ws + WT_BYTES);

    prep_weights<<<LAYERS, 256, 0, stream>>>(w_ih, b_ih, b_hh, wt, bsum);
    lstm_stack<<<NBLOCKS, 256, 0, stream>>>(tokens, emb, wt, bsum, w_out, b_out, out);
}
